// RelationGraphConvNet_53455162966133
// MI455X (gfx1250) — matrix operand order repaired and checked by reading, not compiled, not run on hardware
//
#include <hip/hip_runtime.h>
#include <math.h>

typedef __attribute__((ext_vector_type(16))) _Float16 v16h;
typedef __attribute__((ext_vector_type(8)))  float    v8f;

#define RREL   3
#define NBASIS 5
#define HDIM   16
#define CDIM   8

// ---------------------------------------------------------------------------
// Build f16 weight tables, zero-padded and TRANSPOSED to [slot][N=16][K=32]
// so each lane's WMMA B-fragment (col N, K=kbB..kbB+15) is one contiguous,
// 32B-aligned 32-byte run -> single v16h load.
//   slot r<3 : W[r] = sum_b comp[r,b] * basis[b]   (relation weights)
//   slot 3   : root weight
// W1 table: K(in)=0..2 valid, N(out)=0..15 valid.
// W2 table: K(in)=0..15 valid, N(out)=0..7 valid.
// ---------------------------------------------------------------------------
__global__ void prep_weights(const float* __restrict__ basis1, const float* __restrict__ comp1,
                             const float* __restrict__ root1,
                             const float* __restrict__ basis2, const float* __restrict__ comp2,
                             const float* __restrict__ root2,
                             _Float16* __restrict__ W1t, _Float16* __restrict__ W2t)
{
    for (int t = threadIdx.x; t < 2 * 4 * 16 * 32; t += blockDim.x) {
        int which = t / 2048;
        int idx   = t % 2048;
        int r     = idx / 512;          // slot
        int Nn    = (idx % 512) / 32;   // output column
        int K     = idx % 32;           // input row
        float v = 0.0f;
        if (which == 0) {                    // layer 1: in=3, out=16
            if (K < 3) {
                if (r < RREL) {
                    for (int b = 0; b < NBASIS; ++b)
                        v += comp1[r * NBASIS + b] * basis1[(b * 3 + K) * HDIM + Nn];
                } else {
                    v = root1[K * HDIM + Nn];
                }
            }
            W1t[idx] = (_Float16)v;
        } else {                             // layer 2: in=16, out=8
            if (K < HDIM && Nn < CDIM) {
                if (r < RREL) {
                    for (int b = 0; b < NBASIS; ++b)
                        v += comp2[r * NBASIS + b] * basis2[(b * HDIM + K) * CDIM + Nn];
                } else {
                    v = root2[K * CDIM + Nn];
                }
            }
            W2t[idx] = (_Float16)v;
        }
    }
}

// ---------------------------------------------------------------------------
// Layer-1 node transforms: one wave per 16-node tile, 4 WMMAs per tile
// (3 relations + root). xw1[r][n][16], rootx[n][16].
// ---------------------------------------------------------------------------
__global__ void xw1_wmma(const float* __restrict__ x, const _Float16* __restrict__ W1t,
                         float* __restrict__ xw1, float* __restrict__ rootx, int nN)
{
    int lane   = threadIdx.x & 31;
    int wave   = threadIdx.x >> 5;
    int tile   = blockIdx.x * (blockDim.x >> 5) + wave;
    int ntiles = (nN + 15) >> 4;
    if (tile >= ntiles) return;              // uniform per wave: EXEC stays all-ones

    bool fulltile = (tile * 16 + 16) <= nN;  // wave-uniform fast path

    int m    = lane & 15;
    int node = tile * 16 + m;
    if (node >= nN) node = nN - 1;           // clamp loads (ragged tile only)
    int kb   = (lane < 16) ? 0 : 16;
    const float* xr = x + (size_t)node * 3;

    // A fragment: 16x32 f16, K>=3 zero-padded
    v16h a = {};
#pragma unroll
    for (int j = 0; j < 16; ++j) {
        int K = kb + j;
        if (K < 3) a[j] = (_Float16)xr[K];
    }

    int Nc  = lane & 15;
    int kbB = (lane < 16) ? 0 : 16;
    int rowbase = tile * 16 + ((lane < 16) ? 0 : 8);

#pragma unroll
    for (int r = 0; r < 4; ++r) {
        // transposed table: lane's 16 B values are contiguous -> one v16h load
        v16h b = *(const v16h*)(W1t + r * 512 + Nc * 32 + kbB);
        v8f c = {};
        c = __builtin_amdgcn_wmma_f32_16x16x32_f16(false, a, false, b,
                                                   (short)0, c, false, false);
        float* obuf = (r < RREL) ? (xw1 + (size_t)r * nN * HDIM) : rootx;
        if (fulltile) {
#pragma unroll
            for (int i = 0; i < 8; ++i)
                obuf[(size_t)(rowbase + i) * HDIM + Nc] = c[i];
        } else {
#pragma unroll
            for (int i = 0; i < 8; ++i) {
                int row = rowbase + i;
                if (row < nN) obuf[(size_t)row * HDIM + Nc] = c[i];
            }
        }
    }
}

// ---------------------------------------------------------------------------
// Layer-2 node transforms: A = h tile (16x16 of K=32), B = W2 (16x8 padded).
// xw2[r][n][8], rooth[n][8].
// ---------------------------------------------------------------------------
__global__ void xw2_wmma(const float* __restrict__ h, const _Float16* __restrict__ W2t,
                         float* __restrict__ xw2, float* __restrict__ rooth, int nN)
{
    int lane   = threadIdx.x & 31;
    int wave   = threadIdx.x >> 5;
    int tile   = blockIdx.x * (blockDim.x >> 5) + wave;
    int ntiles = (nN + 15) >> 4;
    if (tile >= ntiles) return;

    bool fulltile = (tile * 16 + 16) <= nN;

    int m    = lane & 15;
    int node = tile * 16 + m;
    if (node >= nN) node = nN - 1;
    int kb   = 0;
    const float* hr = h + (size_t)node * HDIM + kb;

    v16h a = {};
#pragma unroll
    for (int j = 0; j < 16; ++j) if (lane < 16) a[j] = (_Float16)hr[j];   // K = 16*(lane/16)+j; K>=16 zero

    int Nc  = lane & 15;
    int kbB = (lane < 16) ? 0 : 16;
    int rowbase = tile * 16 + ((lane < 16) ? 0 : 8);
    bool colok = (Nc < CDIM);

#pragma unroll
    for (int r = 0; r < 4; ++r) {
        v16h b = *(const v16h*)(W2t + r * 512 + Nc * 32 + kbB);
        v8f c = {};
        c = __builtin_amdgcn_wmma_f32_16x16x32_f16(false, a, false, b,
                                                   (short)0, c, false, false);
        float* obuf = (r < RREL) ? (xw2 + (size_t)r * nN * CDIM) : rooth;
        if (colok) {
            if (fulltile) {
#pragma unroll
                for (int i = 0; i < 8; ++i)
                    obuf[(size_t)(rowbase + i) * CDIM + Nc] = c[i];
            } else {
#pragma unroll
                for (int i = 0; i < 8; ++i) {
                    int row = rowbase + i;
                    if (row < nN) obuf[(size_t)row * CDIM + Nc] = c[i];
                }
            }
        }
    }
}

// ---------------------------------------------------------------------------
// Edge scatter, layer 1: per edge, add message row (16 f32) into
// sum1[rel][dst][*] and bump cnt[rel][dst]. global_atomic_add_f32 path.
// ---------------------------------------------------------------------------
__global__ void scatter1(const int* __restrict__ srcp, const int* __restrict__ dstp,
                         const int* __restrict__ etp, const float* __restrict__ xw1,
                         float* __restrict__ sum1, float* __restrict__ cnt,
                         int nE, int nN)
{
    int e      = blockIdx.x * blockDim.x + threadIdx.x;
    int stride = gridDim.x * blockDim.x;
    for (; e < nE; e += stride) {
        int r = etp[e], s = srcp[e], d = dstp[e];
        const float4* mi = (const float4*)(xw1 + ((size_t)r * nN + s) * HDIM);
        float4 m0 = mi[0], m1 = mi[1], m2 = mi[2], m3 = mi[3];
        float* o = sum1 + ((size_t)r * nN + d) * HDIM;
        atomicAdd(o + 0,  m0.x); atomicAdd(o + 1,  m0.y);
        atomicAdd(o + 2,  m0.z); atomicAdd(o + 3,  m0.w);
        atomicAdd(o + 4,  m1.x); atomicAdd(o + 5,  m1.y);
        atomicAdd(o + 6,  m1.z); atomicAdd(o + 7,  m1.w);
        atomicAdd(o + 8,  m2.x); atomicAdd(o + 9,  m2.y);
        atomicAdd(o + 10, m2.z); atomicAdd(o + 11, m2.w);
        atomicAdd(o + 12, m3.x); atomicAdd(o + 13, m3.y);
        atomicAdd(o + 14, m3.z); atomicAdd(o + 15, m3.w);
        atomicAdd(cnt + (size_t)r * nN + d, 1.0f);
    }
}

// ---------------------------------------------------------------------------
// Edge scatter, layer 2: 8-wide messages; counts reused from layer 1.
// ---------------------------------------------------------------------------
__global__ void scatter2(const int* __restrict__ srcp, const int* __restrict__ dstp,
                         const int* __restrict__ etp, const float* __restrict__ xw2,
                         float* __restrict__ sum2, int nE, int nN)
{
    int e      = blockIdx.x * blockDim.x + threadIdx.x;
    int stride = gridDim.x * blockDim.x;
    for (; e < nE; e += stride) {
        int r = etp[e], s = srcp[e], d = dstp[e];
        const float4* mi = (const float4*)(xw2 + ((size_t)r * nN + s) * CDIM);
        float4 m0 = mi[0], m1 = mi[1];
        float* o = sum2 + ((size_t)r * nN + d) * CDIM;
        atomicAdd(o + 0, m0.x); atomicAdd(o + 1, m0.y);
        atomicAdd(o + 2, m0.z); atomicAdd(o + 3, m0.w);
        atomicAdd(o + 4, m1.x); atomicAdd(o + 5, m1.y);
        atomicAdd(o + 6, m1.z); atomicAdd(o + 7, m1.w);
    }
}

// ---------------------------------------------------------------------------
// Finalize layer 1: mean per relation, sum relations, + root + bias, ReLU.
// ---------------------------------------------------------------------------
__global__ void finalize1(const float* __restrict__ sum1, const float* __restrict__ cnt,
                          const float* __restrict__ rootx, const float* __restrict__ bias1,
                          float* __restrict__ h, int nN)
{
    int n = blockIdx.x * blockDim.x + threadIdx.x;
    if (n >= nN) return;
    float acc[HDIM];
#pragma unroll
    for (int k = 0; k < HDIM; ++k) acc[k] = rootx[(size_t)n * HDIM + k] + bias1[k];
#pragma unroll
    for (int r = 0; r < RREL; ++r) {
        float c   = cnt[(size_t)r * nN + n];
        float inv = 1.0f / fmaxf(c, 1.0f);
        const float* sp = sum1 + ((size_t)r * nN + n) * HDIM;
#pragma unroll
        for (int k = 0; k < HDIM; ++k) acc[k] += sp[k] * inv;
    }
#pragma unroll
    for (int k = 0; k < HDIM; ++k) h[(size_t)n * HDIM + k] = fmaxf(acc[k], 0.0f);
}

// ---------------------------------------------------------------------------
// Finalize layer 2 + log_softmax over 8 classes.
// ---------------------------------------------------------------------------
__global__ void finalize2(const float* __restrict__ sum2, const float* __restrict__ cnt,
                          const float* __restrict__ rooth, const float* __restrict__ bias2,
                          float* __restrict__ out, int nN)
{
    int n = blockIdx.x * blockDim.x + threadIdx.x;
    if (n >= nN) return;
    float o[CDIM];
#pragma unroll
    for (int c = 0; c < CDIM; ++c) o[c] = rooth[(size_t)n * CDIM + c] + bias2[c];
#pragma unroll
    for (int r = 0; r < RREL; ++r) {
        float cn  = cnt[(size_t)r * nN + n];
        float inv = 1.0f / fmaxf(cn, 1.0f);
        const float* sp = sum2 + ((size_t)r * nN + n) * CDIM;
#pragma unroll
        for (int c = 0; c < CDIM; ++c) o[c] += sp[c] * inv;
    }
    float mx = o[0];
#pragma unroll
    for (int c = 1; c < CDIM; ++c) mx = fmaxf(mx, o[c]);
    float s = 0.0f;
#pragma unroll
    for (int c = 0; c < CDIM; ++c) s += expf(o[c] - mx);
    float l = logf(s);
#pragma unroll
    for (int c = 0; c < CDIM; ++c) out[(size_t)n * CDIM + c] = o[c] - mx - l;
}

// ---------------------------------------------------------------------------
extern "C" void kernel_launch(void* const* d_in, const int* in_sizes, int n_in,
                              void* d_out, int out_size, void* d_ws, size_t ws_size,
                              hipStream_t stream)
{
    const float* x      = (const float*)d_in[0];
    const int*   ei     = (const int*)  d_in[1];
    const int*   et     = (const int*)  d_in[2];
    const float* basis1 = (const float*)d_in[3];
    const float* comp1  = (const float*)d_in[4];
    const float* root1  = (const float*)d_in[5];
    const float* bias1  = (const float*)d_in[6];
    const float* basis2 = (const float*)d_in[7];
    const float* comp2  = (const float*)d_in[8];
    const float* root2  = (const float*)d_in[9];
    const float* bias2  = (const float*)d_in[10];
    float* out = (float*)d_out;

    const int nN = in_sizes[0] / 3;
    const int nE = in_sizes[2];
    const int* srcp = ei;
    const int* dstp = ei + nE;

    // ---- carve workspace (256B-aligned chunks) ----
    char* w = (char*)d_ws;
    size_t off = 0;
    auto carve = [&](size_t bytes) -> char* {
        char* p = w + off;
        off += (bytes + 255) & ~(size_t)255;
        return p;
    };
    _Float16* W1t   = (_Float16*)carve(4 * 16 * 32 * sizeof(_Float16));
    _Float16* W2t   = (_Float16*)carve(4 * 16 * 32 * sizeof(_Float16));
    float*    xw1   = (float*)carve((size_t)RREL * nN * HDIM * sizeof(float));
    float*    rootx = (float*)carve((size_t)nN * HDIM * sizeof(float));
    float*    sum1  = (float*)carve((size_t)RREL * nN * HDIM * sizeof(float));
    float*    cnt   = (float*)carve((size_t)RREL * nN * sizeof(float));
    float*    hbuf  = (float*)carve((size_t)nN * HDIM * sizeof(float));
    float*    xw2   = (float*)carve((size_t)RREL * nN * CDIM * sizeof(float));
    float*    rooth = (float*)carve((size_t)nN * CDIM * sizeof(float));
    float*    sum2  = (float*)carve((size_t)RREL * nN * CDIM * sizeof(float));
    (void)ws_size; (void)n_in; (void)out_size;

    // ---- zero accumulators (graph-capture-legal stream memsets) ----
    hipMemsetAsync(sum1, 0, (size_t)RREL * nN * HDIM * sizeof(float), stream);
    hipMemsetAsync(cnt,  0, (size_t)RREL * nN * sizeof(float), stream);
    hipMemsetAsync(sum2, 0, (size_t)RREL * nN * CDIM * sizeof(float), stream);

    // ---- pipeline ----
    prep_weights<<<1, 256, 0, stream>>>(basis1, comp1, root1, basis2, comp2, root2, W1t, W2t);

    int ntiles  = (nN + 15) / 16;
    int wavesPB = 256 / 32;
    int gTiles  = (ntiles + wavesPB - 1) / wavesPB;

    xw1_wmma<<<gTiles, 256, 0, stream>>>(x, W1t, xw1, rootx, nN);
    scatter1<<<(nE + 255) / 256, 256, 0, stream>>>(srcp, dstp, et, xw1, sum1, cnt, nE, nN);
    finalize1<<<(nN + 255) / 256, 256, 0, stream>>>(sum1, cnt, rootx, bias1, hbuf, nN);

    xw2_wmma<<<gTiles, 256, 0, stream>>>(hbuf, W2t, xw2, rooth, nN);
    scatter2<<<(nE + 255) / 256, 256, 0, stream>>>(srcp, dstp, et, xw2, sum2, nE, nN);
    finalize2<<<(nN + 255) / 256, 256, 0, stream>>>(sum2, cnt, rooth, bias2, out, nN);
}
